// SequentialModeCoupling_91164975825303
// MI455X (gfx1250) — matrix operand order repaired and checked by reading, not compiled, not run on hardware
//
#include <hip/hip_runtime.h>
#include <hip/hip_bf16.h>
#include <math.h>

// ---------------------------------------------------------------------------
// SequentialModeCoupling on MI455X (gfx1250, wave32)
//   - complex GEMMs via v_wmma_f32_16x16x32_bf16 (f32 accumulate)
//   - gemm_score stages tiles LDS-side with global_load_async_to_lds_b128
//     + s_wait_asynccnt (per-wave ASYNCcnt double buffering)
// ---------------------------------------------------------------------------

#define BATCH   8
#define S_ELEMS (1 << 20)          // 128*128*64
#define EPSF    1e-8f

typedef __attribute__((ext_vector_type(16))) __bf16       v16bf;
typedef __attribute__((ext_vector_type(8)))  float        v8f;
typedef __attribute__((ext_vector_type(4)))  unsigned int v4u;

union Frag { unsigned int u[8]; v4u q[2]; v16bf v; };

__device__ __forceinline__ v8f wmma_bf16(v16bf a, v16bf b, v8f c) {
    return __builtin_amdgcn_wmma_f32_16x16x32_bf16(
        false, a, false, b, (short)0, c, false, false);
}

// per-mode compile-time traits
template<int MODE> struct MT;
template<> struct MT<0> { static constexpr int D = 128, LR = 13, SPLITK = 8;  };
template<> struct MT<1> { static constexpr int D = 128, LR = 13, SPLITK = 8;  };
template<> struct MT<2> { static constexpr int D = 64,  LR = 14, SPLITK = 32; };

// flat index into (B,128,128,64) state for element (b, row i, col k) of u_mode
template<int MODE>
__device__ __forceinline__ int xindex(int b, int i, int k) {
    int base = b << 20;
    if constexpr (MODE == 0) return base + (i << 13) + k;
    else if constexpr (MODE == 1) return base + ((k >> 6) << 13) + (i << 6) + (k & 63);
    else return base + (k << 6) + i;
}

// --------------------------- utility kernels -------------------------------

__global__ void convert_w_kernel(const float* wr, const float* wi,
                                 __bf16* br, __bf16* bi, int n) {
    int i = blockIdx.x * blockDim.x + threadIdx.x;
    if (i < n) { br[i] = (__bf16)wr[i]; bi[i] = (__bf16)wi[i]; }
}

// state (f32, mode-strided) -> u (B,d,r) bf16 row-major AND uT (B,r,d) bf16
template<int MODE>
__global__ void unfold_kernel(const float* __restrict__ xr,
                              const float* __restrict__ xi,
                              __bf16* __restrict__ ur,  __bf16* __restrict__ ui,
                              __bf16* __restrict__ utr, __bf16* __restrict__ uti) {
    constexpr int D = MT<MODE>::D, LR = MT<MODE>::LR, R = 1 << LR;
    int n = blockIdx.x * blockDim.x + threadIdx.x;
    if (n >= BATCH * S_ELEMS) return;
    int b   = n >> 20;
    int rem = n & (S_ELEMS - 1);
    int i   = rem >> LR;
    int k   = rem & (R - 1);
    int xi_ = xindex<MODE>(b, i, k);
    __bf16 vr = (__bf16)xr[xi_];
    __bf16 vi = (__bf16)xi[xi_];
    ur[n] = vr;  ui[n] = vi;
    int tidx = (b << 20) + k * D + i;
    utr[tidx] = vr;  uti[tidx] = vi;
}

// ---------------- complex GEMM: out = A(d,d) * u(d,r) ----------------------
// OUTX=0: A = shared weight, store bf16 left.  OUTX=1: A = per-batch routing
// matrix, scatter f32 straight back into the x layout (fold fused).
template<int MODE, int OUTX>
__global__ void gemm_aun_kernel(const __bf16* __restrict__ Ar,
                                const __bf16* __restrict__ Ai,
                                const __bf16* __restrict__ uTr,
                                const __bf16* __restrict__ uTi,
                                __bf16* __restrict__ Lr, __bf16* __restrict__ Li,
                                float* __restrict__ Xr, float* __restrict__ Xi) {
    constexpr int D = MT<MODE>::D, LR = MT<MODE>::LR, R = 1 << LR;
    constexpr int TILESN = R >> 4;
    constexpr int TILESPB = (D >> 4) * TILESN;     // 4096 for every mode

    int wid  = blockIdx.x * (blockDim.x >> 5) + (threadIdx.x >> 5);
    int lane = threadIdx.x & 31;
    int l15 = lane & 15, lhalf = lane >> 4;

    int b  = wid / TILESPB;
    int t  = wid - b * TILESPB;
    int tm = t / TILESN;
    int tn = t - tm * TILESN;
    int m0 = tm << 4, n0 = tn << 4;

    const __bf16* ar = Ar + (OUTX ? b * D * D : 0);
    const __bf16* ai = Ai + (OUTX ? b * D * D : 0);
    const __bf16* btr = uTr + (b << 20) + (n0 + l15) * D;   // uT row = column n
    const __bf16* bti = uTi + (b << 20) + (n0 + l15) * D;

    int arow = m0 + l15;
    v8f accR = {}; v8f accI = {};

    for (int k0 = 0; k0 < D; k0 += 32) {
        if (k0 + 32 < D) {
            __builtin_prefetch(btr + k0 + 32, 0, 0);
            __builtin_prefetch(bti + k0 + 32, 0, 0);
        }
        Frag a_r, a_i, a_in, b_r, b_i;
        // A fragment: two 16B loads (elements k0+16*lhalf .. +15)
        const __bf16* arp = ar + arow * D + k0 + 16 * lhalf;
        const __bf16* aip = ai + arow * D + k0 + 16 * lhalf;
        a_r.q[0] = *(const v4u*)(arp);      a_r.q[1] = *(const v4u*)(arp + 8);
        a_i.q[0] = *(const v4u*)(aip);      a_i.q[1] = *(const v4u*)(aip + 8);
#pragma unroll
        for (int v = 0; v < 8; ++v) a_in.u[v] = a_i.u[v] ^ 0x80008000u;
        // B fragment from uT: 16 contiguous bf16 (K-pairs) per lane
        const __bf16* brp = btr + k0 + 16 * lhalf;
        const __bf16* bip = bti + k0 + 16 * lhalf;
        b_r.q[0] = *(const v4u*)(brp);      b_r.q[1] = *(const v4u*)(brp + 8);
        b_i.q[0] = *(const v4u*)(bip);      b_i.q[1] = *(const v4u*)(bip + 8);

        accR = wmma_bf16(a_r.v,  b_r.v, accR);
        accR = wmma_bf16(a_in.v, b_i.v, accR);
        accI = wmma_bf16(a_r.v,  b_i.v, accI);
        accI = wmma_bf16(a_i.v,  b_r.v, accI);
    }

    if constexpr (!OUTX) {
#pragma unroll
        for (int vc = 0; vc < 8; ++vc) {
            int row = m0 + vc + 8 * lhalf;
            int col = n0 + l15;
            int idx = (b << 20) + (row << LR) + col;
            Lr[idx] = (__bf16)accR[vc];
            Li[idx] = (__bf16)accI[vc];
        }
    } else {
#pragma unroll
        for (int vc = 0; vc < 8; ++vc) {
            int row = m0 + vc + 8 * lhalf;
            int col = n0 + l15;
            int xi_ = xindex<MODE>(b, row, col);
            Xr[xi_] = accR[vc];
            Xi[xi_] = accI[vc];
        }
    }
}

// -------- complex GEMM: score(d,d) = left(d,r) * u(d,r)^T, split-K ---------
// Tiles staged into per-wave LDS with global_load_async_to_lds_b128, double-
// buffered on the per-wave ASYNCcnt; deterministic partial-sum epilogue.
template<int MODE>
__global__ void gemm_score_kernel(const __bf16* __restrict__ Lr,
                                  const __bf16* __restrict__ Li,
                                  const __bf16* __restrict__ Ur,
                                  const __bf16* __restrict__ Ui,
                                  float* __restrict__ Pr, float* __restrict__ Pi) {
    constexpr int D = MT<MODE>::D, LR = MT<MODE>::LR, R = 1 << LR;
    constexpr int SPLITK = MT<MODE>::SPLITK;
    constexpr int TILESD = D >> 4;
    constexpr int PERB   = TILESD * TILESD * SPLITK;   // 512 for every mode
    constexpr int KCHUNK = R / SPLITK;
    constexpr int STEPS  = KCHUNK / 32;

    // [wave][stage][tile: Are,Aim,Bre,Bim][16 rows x 32 bf16 = 1KB]
    __shared__ unsigned char shbuf[4][2][4][1024];

    int w    = threadIdx.x >> 5;
    int lane = threadIdx.x & 31;
    int l15 = lane & 15, lhalf = lane >> 4;

    int wid  = blockIdx.x * (blockDim.x >> 5) + w;
    int b    = wid / PERB;
    int t2   = wid - b * PERB;
    int tile = t2 / SPLITK;
    int ks   = t2 - tile * SPLITK;
    int tm = tile / TILESD, tn = tile - tm * TILESD;
    int m0 = tm << 4, n0 = tn << 4;
    int kbeg = ks * KCHUNK;

    const __bf16* g0 = Lr + (b << 20) + m0 * R;   // A real tile rows
    const __bf16* g1 = Li + (b << 20) + m0 * R;
    const __bf16* g2 = Ur + (b << 20) + n0 * R;   // B tile = u rows n0..n0+15
    const __bf16* g3 = Ui + (b << 20) + n0 * R;

    auto issue = [&](int s, int k0) {
        const __bf16* gb[4] = { g0 + k0, g1 + k0, g2 + k0, g3 + k0 };
#pragma unroll
        for (int ti = 0; ti < 4; ++ti) {
            unsigned lds0 = (unsigned)(uintptr_t)(void*)&shbuf[w][s][ti][0];
#pragma unroll
            for (int h = 0; h < 2; ++h) {
                int c = h * 32 + lane;              // 64 x 16B chunks per tile
                int row = c >> 2, cir = c & 3;
                unsigned long long ga =
                    (unsigned long long)(uintptr_t)(gb[ti] + row * R + cir * 8);
                unsigned la = lds0 + c * 16;
                asm volatile("global_load_async_to_lds_b128 %0, %1, off"
                             :: "v"(la), "v"(ga) : "memory");
            }
        }
    };

    issue(0, kbeg);
    v8f accR = {}; v8f accI = {};

#pragma unroll 1
    for (int t = 0; t < STEPS; ++t) {
        int s = t & 1;
        if (t + 1 < STEPS) {
            issue(s ^ 1, kbeg + (t + 1) * 32);
            asm volatile("s_wait_asynccnt 0x8" ::: "memory");
        } else {
            asm volatile("s_wait_asynccnt 0x0" ::: "memory");
        }
        const unsigned int* sA0 = (const unsigned int*)&shbuf[w][s][0][0];
        const unsigned int* sA1 = (const unsigned int*)&shbuf[w][s][1][0];
        const unsigned int* sB0 = (const unsigned int*)&shbuf[w][s][2][0];
        const unsigned int* sB1 = (const unsigned int*)&shbuf[w][s][3][0];

        Frag a_r, a_i, a_in, b_r, b_i;
        const v4u* pAr = (const v4u*)(sA0 + l15 * 16 + 8 * lhalf);
        const v4u* pAi = (const v4u*)(sA1 + l15 * 16 + 8 * lhalf);
        a_r.q[0] = pAr[0];  a_r.q[1] = pAr[1];      // dwords +0..7
        a_i.q[0] = pAi[0];  a_i.q[1] = pAi[1];
#pragma unroll
        for (int v = 0; v < 8; ++v) a_in.u[v] = a_i.u[v] ^ 0x80008000u;
        const v4u* pBr = (const v4u*)(sB0 + l15 * 16 + 8 * lhalf);
        const v4u* pBi = (const v4u*)(sB1 + l15 * 16 + 8 * lhalf);
        b_r.q[0] = pBr[0];  b_r.q[1] = pBr[1];      // dwords +0..7
        b_i.q[0] = pBi[0];  b_i.q[1] = pBi[1];

        accR = wmma_bf16(a_r.v,  b_r.v, accR);
        accR = wmma_bf16(a_in.v, b_i.v, accR);
        accI = wmma_bf16(a_r.v,  b_i.v, accI);
        accI = wmma_bf16(a_i.v,  b_r.v, accI);
    }

    // deterministic partials: P[ks][b][row][col]
#pragma unroll
    for (int vc = 0; vc < 8; ++vc) {
        int row = m0 + vc + 8 * lhalf;
        int col = n0 + l15;
        int idx = ((ks * BATCH + b) * D + row) * D + col;
        Pr[idx] = accR[vc];
        Pi[idx] = accI[vc];
    }
}

__global__ void reduce_score_kernel(const float* __restrict__ pr,
                                    const float* __restrict__ pi,
                                    float* __restrict__ sr, float* __restrict__ si,
                                    int n, int splitk) {
    int i = blockIdx.x * blockDim.x + threadIdx.x;
    if (i >= n) return;
    float ar = 0.0f, ai = 0.0f;
    for (int s = 0; s < splitk; ++s) { ar += pr[s * n + i]; ai += pi[s * n + i]; }
    sr[i] = ar;  si[i] = ai;
}

// --------- routing = softmax(|score|/scale); matrix = routing*phase --------
__global__ void softmax_phase_kernel(const float* __restrict__ Sr,
                                     const float* __restrict__ Si,
                                     __bf16* __restrict__ Mr,
                                     __bf16* __restrict__ Mi,
                                     const float* __restrict__ log_tau,
                                     int d, float scale_mult) {
    __shared__ float sh[128];
    int j = threadIdx.x;
    int base = blockIdx.x * d;               // blockIdx = b*d + i
    float re = Sr[base + j], im = Si[base + j];
    float mag = sqrtf(re * re + im * im);
    float tau = fmaxf(__expf(log_tau[0]), EPSF);
    float val = mag / (tau * scale_mult);

    sh[j] = val; __syncthreads();
    for (int s = d >> 1; s > 0; s >>= 1) {
        if (j < s) sh[j] = fmaxf(sh[j], sh[j + s]);
        __syncthreads();
    }
    float vmax = sh[0]; __syncthreads();
    float ex = __expf(val - vmax);
    sh[j] = ex; __syncthreads();
    for (int s = d >> 1; s > 0; s >>= 1) {
        if (j < s) sh[j] += sh[j + s];
        __syncthreads();
    }
    float routing = ex / sh[0];

    float safe = fmaxf(mag, EPSF);
    float pr = (mag > EPSF) ? re / safe : 1.0f;
    float pi = (mag > EPSF) ? im / safe : 0.0f;
    Mr[base + j] = (__bf16)(routing * pr);
    Mi[base + j] = (__bf16)(routing * pi);
}

// ------------------------------- launcher ----------------------------------

struct Bufs {
    __bf16 *u_r, *u_i, *ut_r, *ut_i, *l_r, *l_i, *m_r, *m_i, *w_r, *w_i;
    float  *p_r, *p_i, *s_r, *s_i;
};

template<int MODE>
static void run_mode(const Bufs& B_, const float* srcR, const float* srcI,
                     const float* wrf, const float* wif, const float* log_tau,
                     float* outR, float* outI, hipStream_t stream) {
    constexpr int D = MT<MODE>::D, SPLITK = MT<MODE>::SPLITK;
    const size_t BS = (size_t)BATCH * S_ELEMS;
    float scale_mult = sqrtf((float)S_ELEMS / (float)D);

    convert_w_kernel<<<(D * D + 255) / 256, 256, 0, stream>>>(
        wrf, wif, B_.w_r, B_.w_i, D * D);

    unfold_kernel<MODE><<<(int)(BS / 256), 256, 0, stream>>>(
        srcR, srcI, B_.u_r, B_.u_i, B_.ut_r, B_.ut_i);

    // phase 1: left = W * u   (B*4096 tiles, 4 waves/block)
    gemm_aun_kernel<MODE, 0><<<BATCH * 4096 / 4, 128, 0, stream>>>(
        B_.w_r, B_.w_i, B_.ut_r, B_.ut_i, B_.l_r, B_.l_i, nullptr, nullptr);

    // phase 2: score = left * u^T  (async-LDS staged, split-K partials)
    gemm_score_kernel<MODE><<<BATCH * 512 / 4, 128, 0, stream>>>(
        B_.l_r, B_.l_i, B_.u_r, B_.u_i, B_.p_r, B_.p_i);
    int n = BATCH * D * D;
    reduce_score_kernel<<<(n + 255) / 256, 256, 0, stream>>>(
        B_.p_r, B_.p_i, B_.s_r, B_.s_i, n, SPLITK);

    softmax_phase_kernel<<<BATCH * D, D, 0, stream>>>(
        B_.s_r, B_.s_i, B_.m_r, B_.m_i, log_tau, D, scale_mult);

    // phase 3: mixed = matrix * u, folded straight back into state layout
    gemm_aun_kernel<MODE, 1><<<BATCH * 4096 / 4, 128, 0, stream>>>(
        B_.m_r, B_.m_i, B_.ut_r, B_.ut_i, nullptr, nullptr, outR, outI);
}

extern "C" void kernel_launch(void* const* d_in, const int* in_sizes, int n_in,
                              void* d_out, int out_size, void* d_ws, size_t ws_size,
                              hipStream_t stream) {
    (void)in_sizes; (void)n_in; (void)out_size; (void)ws_size;
    const size_t BS = (size_t)BATCH * S_ELEMS;

    const float* x_r = (const float*)d_in[0];
    const float* x_i = (const float*)d_in[1];
    const float* log_tau = (const float*)d_in[8];

    float* outR = (float*)d_out;
    float* outI = outR + BS;

    char* ws = (char*)d_ws;
    auto carve = [&](size_t bytes) -> void* {
        void* p = (void*)ws;
        ws += (bytes + 255) & ~(size_t)255;
        return p;
    };
    Bufs B_;
    B_.u_r  = (__bf16*)carve(BS * 2);
    B_.u_i  = (__bf16*)carve(BS * 2);
    B_.ut_r = (__bf16*)carve(BS * 2);
    B_.ut_i = (__bf16*)carve(BS * 2);
    B_.l_r  = (__bf16*)carve(BS * 2);
    B_.l_i  = (__bf16*)carve(BS * 2);
    B_.p_r  = (float*)carve((size_t)32 * BATCH * 128 * 128 * 4 / 8);  // 2 MB
    B_.p_i  = (float*)carve((size_t)32 * BATCH * 128 * 128 * 4 / 8);
    B_.s_r  = (float*)carve((size_t)BATCH * 128 * 128 * 4);
    B_.s_i  = (float*)carve((size_t)BATCH * 128 * 128 * 4);
    B_.m_r  = (__bf16*)carve((size_t)BATCH * 128 * 128 * 2);
    B_.m_i  = (__bf16*)carve((size_t)BATCH * 128 * 128 * 2);
    B_.w_r  = (__bf16*)carve(128 * 128 * 2);
    B_.w_i  = (__bf16*)carve(128 * 128 * 2);

    run_mode<0>(B_, x_r, x_i, (const float*)d_in[2], (const float*)d_in[3],
                log_tau, outR, outI, stream);
    run_mode<1>(B_, outR, outI, (const float*)d_in[4], (const float*)d_in[5],
                log_tau, outR, outI, stream);
    run_mode<2>(B_, outR, outI, (const float*)d_in[6], (const float*)d_in[7],
                log_tau, outR, outI, stream);
}
